// PanDrugRegressor_19954418057322
// MI455X (gfx1250) — matrix operand order repaired and checked by reading, not compiled, not run on hardware
//
#include <hip/hip_runtime.h>
#include <math.h>

// ---------------------------------------------------------------------------
// Types for CDNA5 WMMA
// ---------------------------------------------------------------------------
typedef __attribute__((ext_vector_type(16))) _Float16 v16h;
typedef __attribute__((ext_vector_type(8)))  _Float16 v8h;
typedef __attribute__((ext_vector_type(8)))  float    v8f;

#define BATCH 32
#define SEQ   1024
#define BS    (BATCH * SEQ)   // 32768 rows
#define DMODEL 64
#define NHEAD  4
#define DH     16
#define FFN    128
#define EMB    32
#define CCH    64
#define CTX    511

__device__ __forceinline__ float gelu_exact(float v) {
    return 0.5f * v * (1.0f + erff(v * 0.70710678118654752f));
}

__device__ __forceinline__ float wave_sum32(float v) {
    #pragma unroll
    for (int o = 16; o > 0; o >>= 1) v += __shfl_xor(v, o, 32);
    return v;
}

// ---------------------------------------------------------------------------
// f32 -> f16 conversion (weights, activations)
// ---------------------------------------------------------------------------
__global__ void cvt_f16_kernel(const float* __restrict__ src, _Float16* __restrict__ dst, int n) {
    int i = blockIdx.x * blockDim.x + threadIdx.x;
    if (i < n) dst[i] = (_Float16)src[i];
}

// ---------------------------------------------------------------------------
// drug_seq[b, e] = sum_j drug_embed[id[b], j] * dsp_w[e, j]
// ---------------------------------------------------------------------------
__global__ void drugseq_kernel(const int* __restrict__ ids, const float* __restrict__ demb,
                               const float* __restrict__ dspw, float* __restrict__ out) {
    int idx = blockIdx.x * blockDim.x + threadIdx.x;   // BATCH*EMB
    if (idx >= BATCH * EMB) return;
    int e = idx & (EMB - 1);
    int b = idx >> 5;
    const float* de = demb + ids[b] * 32;
    float a = 0.f;
    #pragma unroll
    for (int j = 0; j < 32; j++) a += de[j] * dspw[e * 32 + j];
    out[idx] = a;
}

// ---------------------------------------------------------------------------
// embed: e0[m, e] = nt_embed[tok[m], e] + pos_embed[clamp(pos[m]+CTX), e] + drugseq[b, e]
// ---------------------------------------------------------------------------
__global__ void embed_kernel(const int* __restrict__ tok, const int* __restrict__ pos,
                             const float* __restrict__ ntw, const float* __restrict__ pw,
                             const float* __restrict__ dseq, float* __restrict__ out) {
    int idx = blockIdx.x * blockDim.x + threadIdx.x;   // BS*EMB
    if (idx >= BS * EMB) return;
    int e = idx & (EMB - 1);
    int m = idx >> 5;
    int b = m >> 10;
    int pi = pos[m] + CTX; if (pi < 0) pi = 0;
    out[idx] = ntw[tok[m] * EMB + e] + pw[pi * EMB + e] + dseq[b * EMB + e];
}

// ---------------------------------------------------------------------------
// conv1: [BS,32] -> [BS,64], k=5 pad=2 over s, gelu, LN over 64 channels
// block = 64 threads (one output position), grid = BS
// ---------------------------------------------------------------------------
__global__ void conv1_kernel(const float* __restrict__ in, const float* __restrict__ w,
                             const float* __restrict__ g, const float* __restrict__ bt,
                             float* __restrict__ out) {
    __shared__ float tile[5 * 32];
    __shared__ float red[64];
    __shared__ float stat[2];
    int m = blockIdx.x, b = m >> 10, s = m & (SEQ - 1);
    int t = threadIdx.x;
    for (int i = t; i < 5 * 32; i += 64) {
        int tt = i >> 5, c = i & 31;
        int ss = s + tt - 2;
        tile[i] = (ss >= 0 && ss < SEQ) ? in[((size_t)(b * SEQ + ss)) * 32 + c] : 0.f;
    }
    __syncthreads();
    float acc = 0.f;
    const float* wr = w + t * 160;   // [64][32][5]
    #pragma unroll 1
    for (int tt = 0; tt < 5; tt++)
        for (int ic = 0; ic < 32; ic++)
            acc += wr[ic * 5 + tt] * tile[tt * 32 + ic];
    acc = gelu_exact(acc);
    red[t] = acc; __syncthreads();
    for (int o = 32; o > 0; o >>= 1) { if (t < o) red[t] += red[t + o]; __syncthreads(); }
    if (t == 0) stat[0] = red[0] * (1.f / 64.f);
    __syncthreads();
    float d = acc - stat[0];
    red[t] = d * d; __syncthreads();
    for (int o = 32; o > 0; o >>= 1) { if (t < o) red[t] += red[t + o]; __syncthreads(); }
    if (t == 0) stat[1] = red[0] * (1.f / 64.f);
    __syncthreads();
    out[(size_t)m * 64 + t] = d * rsqrtf(stat[1] + 1e-5f) * g[t] + bt[t];
}

// ---------------------------------------------------------------------------
// conv2: [BS,64] -> [BS,64], k=3 pad=1, gelu, LN over 64 channels
// ---------------------------------------------------------------------------
__global__ void conv2_kernel(const float* __restrict__ in, const float* __restrict__ w,
                             const float* __restrict__ g, const float* __restrict__ bt,
                             float* __restrict__ out) {
    __shared__ float tile[3 * 64];
    __shared__ float red[64];
    __shared__ float stat[2];
    int m = blockIdx.x, b = m >> 10, s = m & (SEQ - 1);
    int t = threadIdx.x;
    for (int i = t; i < 3 * 64; i += 64) {
        int tt = i >> 6, c = i & 63;
        int ss = s + tt - 1;
        tile[i] = (ss >= 0 && ss < SEQ) ? in[((size_t)(b * SEQ + ss)) * 64 + c] : 0.f;
    }
    __syncthreads();
    float acc = 0.f;
    const float* wr = w + t * 192;   // [64][64][3]
    #pragma unroll 1
    for (int tt = 0; tt < 3; tt++)
        for (int ic = 0; ic < 64; ic++)
            acc += wr[ic * 3 + tt] * tile[tt * 64 + ic];
    acc = gelu_exact(acc);
    red[t] = acc; __syncthreads();
    for (int o = 32; o > 0; o >>= 1) { if (t < o) red[t] += red[t + o]; __syncthreads(); }
    if (t == 0) stat[0] = red[0] * (1.f / 64.f);
    __syncthreads();
    float d = acc - stat[0];
    red[t] = d * d; __syncthreads();
    for (int o = 32; o > 0; o >>= 1) { if (t < o) red[t] += red[t + o]; __syncthreads(); }
    if (t == 0) stat[1] = red[0] * (1.f / 64.f);
    __syncthreads();
    out[(size_t)m * 64 + t] = d * rsqrtf(stat[1] + 1e-5f) * g[t] + bt[t];
}

// ---------------------------------------------------------------------------
// LayerNorm over 64 cols, write f16. One wave per row.
// ---------------------------------------------------------------------------
__global__ void ln_f16_kernel(const float* __restrict__ x, const float* __restrict__ g,
                              const float* __restrict__ b, _Float16* __restrict__ out, int rows) {
    int warp = (blockIdx.x * blockDim.x + threadIdx.x) >> 5;
    int lane = threadIdx.x & 31;
    if (warp >= rows) return;
    const float* r = x + (size_t)warp * 64;
    float a0 = r[lane], a1 = r[lane + 32];
    float mu = wave_sum32(a0 + a1) * (1.f / 64.f);
    float d0 = a0 - mu, d1 = a1 - mu;
    float var = wave_sum32(d0 * d0 + d1 * d1) * (1.f / 64.f);
    float inv = rsqrtf(var + 1e-5f);
    out[(size_t)warp * 64 + lane]      = (_Float16)(d0 * inv * g[lane] + b[lane]);
    out[(size_t)warp * 64 + lane + 32] = (_Float16)(d1 * inv * g[lane + 32] + b[lane + 32]);
}

// ---------------------------------------------------------------------------
// WMMA GEMM: out[M,N] = A[M,K](f16) * W[N,K]^T(f16) + bias (+resid) (gelu?)
// One wave computes one 16x16 tile: v_wmma_f32_16x16x32_f16, K-loop step 32.
// A 16x32 layout: lane = m + 16*((k>>3)&1); elem e <-> k = (e>>3)*16 + (e&7) + (lane>>4)*8
// B 32x16 layout (B = W^T): lane = n + 16*((k>>3)&1); elem e <-> k = (e>>3)*16 + (e&7) + (lane>>4)*8
// D 16x16 f32:  lane 0-15: N=lane, VGPR r -> M=r; lanes 16-31: N=lane-16, M=8+r
// ---------------------------------------------------------------------------
#define GF_F16OUT 1
#define GF_GELU   2
#define GF_RESID  4

__global__ void gemm_wmma_kernel(const _Float16* __restrict__ A, const _Float16* __restrict__ W,
                                 const float* __restrict__ bias, const float* __restrict__ resid,
                                 float* __restrict__ outF, _Float16* __restrict__ outH,
                                 int N, int K, int flags) {
    int lane = threadIdx.x;
    int hi   = lane >> 4;
    int l15  = lane & 15;
    int mt   = blockIdx.x << 4;
    int nt   = blockIdx.y << 4;
    v8f acc = {};
    const _Float16* arow = A + (size_t)(mt + l15) * K + hi * 8;
    const _Float16* wrow = W + (size_t)(nt + l15) * K + hi * 8;
    for (int k0 = 0; k0 < K; k0 += 32) {
        v8h a0 = *(const v8h*)(arow + k0);
        v8h a1 = *(const v8h*)(arow + k0 + 16);
        v8h b0 = *(const v8h*)(wrow + k0);
        v8h b1 = *(const v8h*)(wrow + k0 + 16);
        v16h av, bv;
        #pragma unroll
        for (int i = 0; i < 8; i++) {
            av[i] = a0[i]; av[i + 8] = a1[i];
            bv[i] = b0[i]; bv[i + 8] = b1[i];
        }
        acc = __builtin_amdgcn_wmma_f32_16x16x32_f16(false, av, false, bv,
                                                     (short)0, acc, false, false);
    }
    int ncol  = nt + l15;
    int mbase = mt + hi * 8;
    float bvl = bias[ncol];
    #pragma unroll
    for (int r = 0; r < 8; r++) {
        size_t o = (size_t)(mbase + r) * N + ncol;
        float v = acc[r] + bvl;
        if (flags & GF_GELU)  v = gelu_exact(v);
        if (flags & GF_RESID) v += resid[o];
        if (flags & GF_F16OUT) outH[o] = (_Float16)v;
        else                   outF[o] = v;
    }
}

// ---------------------------------------------------------------------------
// Windowed attention (window=7): one thread per (b,s,h).
// qkv rows of 192 halves: [0:64)=q  [64:128)=k  [128:192)=v
// Exactly matches full softmax with -1e9 mask (exp underflows to 0).
// ---------------------------------------------------------------------------
__global__ void win_attn_kernel(const _Float16* __restrict__ qkv, _Float16* __restrict__ out) {
    int idx = blockIdx.x * blockDim.x + threadIdx.x;  // BS*NHEAD
    if (idx >= BS * NHEAD) return;
    int h = idx & 3;
    int m = idx >> 2;
    int s = m & (SEQ - 1);
    int rowbase = m - s;  // b*SEQ
    const _Float16* qr = qkv + (size_t)m * 192 + h * DH;
    float q[DH];
    #pragma unroll
    for (int d = 0; d < DH; d++) q[d] = (float)qr[d];
    float sc[7];
    #pragma unroll
    for (int t = 0; t < 7; t++) {
        int j = s - 3 + t;
        if (j < 0 || j >= SEQ) { sc[t] = -1e30f; continue; }
        const _Float16* kr = qkv + (size_t)(rowbase + j) * 192 + 64 + h * DH;
        float dot = 0.f;
        #pragma unroll
        for (int d = 0; d < DH; d++) dot += q[d] * (float)kr[d];
        sc[t] = dot * 0.25f;   // 1/sqrt(16)
    }
    float mx = sc[0];
    #pragma unroll
    for (int t = 1; t < 7; t++) mx = fmaxf(mx, sc[t]);
    float p[7], sum = 0.f;
    #pragma unroll
    for (int t = 0; t < 7; t++) { p[t] = expf(sc[t] - mx); sum += p[t]; }
    float inv = 1.f / sum;
    float accv[DH];
    #pragma unroll
    for (int d = 0; d < DH; d++) accv[d] = 0.f;
    #pragma unroll
    for (int t = 0; t < 7; t++) {
        int j = s - 3 + t;
        if (j < 0 || j >= SEQ) continue;
        const _Float16* vr = qkv + (size_t)(rowbase + j) * 192 + 128 + h * DH;
        float pt = p[t];
        #pragma unroll
        for (int d = 0; d < DH; d++) accv[d] += pt * (float)vr[d];
    }
    _Float16* orow = out + (size_t)m * 64 + h * DH;
    #pragma unroll
    for (int d = 0; d < DH; d++) orow[d] = (_Float16)(accv[d] * inv);
}

// ---------------------------------------------------------------------------
// Pool query: qp[b,n] = sum_k (base[k] + drug_emb[b]·dpw[k]) * qkvw[n,k] + qkvb[n]
// ---------------------------------------------------------------------------
__global__ void poolq_kernel(const int* __restrict__ ids, const float* __restrict__ demb,
                             const float* __restrict__ baseq, const float* __restrict__ dpw,
                             const float* __restrict__ qkvw, const float* __restrict__ qkvb,
                             float* __restrict__ qp) {
    int idx = blockIdx.x * blockDim.x + threadIdx.x;  // BATCH*64
    if (idx >= BATCH * 64) return;
    int n = idx & 63;
    int b = idx >> 6;
    const float* de = demb + ids[b] * 32;
    float acc = qkvb[n];
    for (int k = 0; k < 64; k++) {
        float qk = baseq[k];
        #pragma unroll
        for (int j = 0; j < 32; j++) qk += de[j] * dpw[k * 32 + j];
        acc += qk * qkvw[n * 64 + k];
    }
    qp[idx] = acc;
}

// ---------------------------------------------------------------------------
// Pool attention: block per (b,h), 256 threads, softmax over S=1024
// kv[BS,128]: cols [0:64)=k, [64:128)=v
// ---------------------------------------------------------------------------
__global__ void pool_attn_kernel(const float* __restrict__ qp, const _Float16* __restrict__ kv,
                                 float* __restrict__ pooled) {
    __shared__ float red[256];
    int bh = blockIdx.x;
    int b = bh >> 2, h = bh & 3;
    int tid = threadIdx.x;
    float q[DH];
    #pragma unroll
    for (int d = 0; d < DH; d++) q[d] = qp[b * 64 + h * DH + d];
    float sc[4];
    #pragma unroll
    for (int i = 0; i < 4; i++) {
        int s = tid + i * 256;
        const _Float16* kr = kv + (size_t)(b * SEQ + s) * 128 + h * DH;
        float dot = 0.f;
        #pragma unroll
        for (int d = 0; d < DH; d++) dot += q[d] * (float)kr[d];
        sc[i] = dot * 0.25f;
    }
    float m = fmaxf(fmaxf(sc[0], sc[1]), fmaxf(sc[2], sc[3]));
    red[tid] = m; __syncthreads();
    for (int o = 128; o > 0; o >>= 1) { if (tid < o) red[tid] = fmaxf(red[tid], red[tid + o]); __syncthreads(); }
    float gmax = red[0]; __syncthreads();
    float p[4], ls = 0.f;
    #pragma unroll
    for (int i = 0; i < 4; i++) { p[i] = expf(sc[i] - gmax); ls += p[i]; }
    red[tid] = ls; __syncthreads();
    for (int o = 128; o > 0; o >>= 1) { if (tid < o) red[tid] += red[tid + o]; __syncthreads(); }
    float inv = 1.f / red[0]; __syncthreads();
    for (int d = 0; d < DH; d++) {
        float part = 0.f;
        #pragma unroll
        for (int i = 0; i < 4; i++) {
            int s = tid + i * 256;
            part += p[i] * (float)kv[(size_t)(b * SEQ + s) * 128 + 64 + h * DH + d];
        }
        red[tid] = part; __syncthreads();
        for (int o = 128; o > 0; o >>= 1) { if (tid < o) red[tid] += red[tid + o]; __syncthreads(); }
        if (tid == 0) pooled[b * 64 + h * DH + d] = red[0] * inv;
        __syncthreads();
    }
}

// ---------------------------------------------------------------------------
// Head: out-proj + LN + gelu MLP -> y[b]. Block per b, 64 threads.
// ---------------------------------------------------------------------------
__global__ void head_kernel(const float* __restrict__ pooled,
                            const float* __restrict__ ow, const float* __restrict__ ob,
                            const float* __restrict__ lng, const float* __restrict__ lnb,
                            const float* __restrict__ h1w, const float* __restrict__ h1b,
                            const float* __restrict__ h2w, const float* __restrict__ h2b,
                            float* __restrict__ out) {
    __shared__ float o[64];
    __shared__ float red[64];
    __shared__ float hh[32];
    __shared__ float stat[2];
    int b = blockIdx.x, t = threadIdx.x;
    float acc = ob[t];
    #pragma unroll
    for (int k = 0; k < 64; k++) acc += pooled[b * 64 + k] * ow[t * 64 + k];
    red[t] = acc; __syncthreads();
    for (int s = 32; s > 0; s >>= 1) { if (t < s) red[t] += red[t + s]; __syncthreads(); }
    if (t == 0) stat[0] = red[0] * (1.f / 64.f);
    __syncthreads();
    float d = acc - stat[0];
    red[t] = d * d; __syncthreads();
    for (int s = 32; s > 0; s >>= 1) { if (t < s) red[t] += red[t + s]; __syncthreads(); }
    if (t == 0) stat[1] = red[0] * (1.f / 64.f);
    __syncthreads();
    o[t] = d * rsqrtf(stat[1] + 1e-5f) * lng[t] + lnb[t];
    __syncthreads();
    if (t < 32) {
        float a = h1b[t];
        #pragma unroll
        for (int k = 0; k < 64; k++) a += o[k] * h1w[t * 64 + k];
        hh[t] = gelu_exact(a);
    }
    __syncthreads();
    if (t < 32) red[t] = hh[t] * h2w[t];
    __syncthreads();
    if (t == 0) {
        float s = 0.f;
        #pragma unroll
        for (int k = 0; k < 32; k++) s += red[k];
        out[b] = s + h2b[0];
    }
}

// ---------------------------------------------------------------------------
// Host orchestration
// ---------------------------------------------------------------------------
extern "C" void kernel_launch(void* const* d_in, const int* in_sizes, int n_in,
                              void* d_out, int out_size, void* d_ws, size_t ws_size,
                              hipStream_t stream) {
    (void)in_sizes; (void)n_in; (void)out_size; (void)ws_size;

    // --- inputs (setup_inputs dict order, depth-first) ---
    const int*   tokens    = (const int*)d_in[0];
    const int*   positions = (const int*)d_in[1];
    const int*   drug_ids  = (const int*)d_in[2];
    const float* nt_embed  = (const float*)d_in[3];
    const float* pos_embed = (const float*)d_in[4];
    const float* drug_emb  = (const float*)d_in[5];
    const float* dsp_w     = (const float*)d_in[6];
    const float* conv1_w   = (const float*)d_in[7];
    const float* ln1_g     = (const float*)d_in[8];
    const float* ln1_b     = (const float*)d_in[9];
    const float* conv2_w   = (const float*)d_in[10];
    const float* ln2_g     = (const float*)d_in[11];
    const float* ln2_b     = (const float*)d_in[12];
    // layers: 12 tensors each, starting at 13 and 25
    const int LBASE[2] = {13, 25};
    // pool: 37..44, head: 45..48
    const float* p_baseq  = (const float*)d_in[37];
    const float* p_dpw    = (const float*)d_in[38];
    const float* p_qkvw   = (const float*)d_in[39];
    const float* p_qkvb   = (const float*)d_in[40];
    const float* p_outw   = (const float*)d_in[41];
    const float* p_outb   = (const float*)d_in[42];
    const float* p_lng    = (const float*)d_in[43];
    const float* p_lnb    = (const float*)d_in[44];
    const float* h_h1w    = (const float*)d_in[45];
    const float* h_h1b    = (const float*)d_in[46];
    const float* h_h2w    = (const float*)d_in[47];
    const float* h_h2b    = (const float*)d_in[48];
    float* y_out = (float*)d_out;

    // --- workspace carve ---
    char* ws = (char*)d_ws;
    size_t off = 0;
    auto take = [&](size_t bytes) -> size_t {
        size_t o = off;
        off += (bytes + 255) & ~(size_t)255;
        return o;
    };
    size_t o_x    = take((size_t)BS * 64 * 4);   // residual stream f32
    size_t o_e0   = take((size_t)BS * 32 * 4);   // embed out
    size_t o_c1   = take((size_t)BS * 64 * 4);   // conv1 out
    size_t o_h16  = take((size_t)BS * 64 * 2);   // LN out f16
    size_t o_b16  = take((size_t)BS * 192 * 2);  // qkv16 / gelu16 / kv16
    size_t o_a16  = take((size_t)BS * 64 * 2);   // attn out f16
    size_t o_x16  = take((size_t)BS * 64 * 2);   // x in f16 (pool kv input)
    size_t o_ds   = take((size_t)BATCH * 32 * 4);
    size_t o_qp   = take((size_t)BATCH * 64 * 4);
    size_t o_pool = take((size_t)BATCH * 64 * 4);
    size_t o_w16  = take((size_t)(2 * 32768 + 8192) * 2);  // f16 weight arena

    float*     x    = (float*)(ws + o_x);
    float*     e0   = (float*)(ws + o_e0);
    float*     c1   = (float*)(ws + o_c1);
    _Float16*  h16  = (_Float16*)(ws + o_h16);
    _Float16*  b16  = (_Float16*)(ws + o_b16);
    _Float16*  a16  = (_Float16*)(ws + o_a16);
    _Float16*  x16  = (_Float16*)(ws + o_x16);
    float*     dseq = (float*)(ws + o_ds);
    float*     qp   = (float*)(ws + o_qp);
    float*     pool = (float*)(ws + o_pool);
    _Float16*  w16  = (_Float16*)(ws + o_w16);

    // per-layer f16 weight slots (halves): qkv 12288, out 4096, ff1 8192, ff2 8192
    auto cvt = [&](const float* src, _Float16* dst, int n) {
        cvt_f16_kernel<<<(n + 255) / 256, 256, 0, stream>>>(src, dst, n);
    };
    _Float16* w_qkv[2]; _Float16* w_out[2]; _Float16* w_ff1[2]; _Float16* w_ff2[2];
    for (int l = 0; l < 2; l++) {
        _Float16* base = w16 + (size_t)l * 32768;
        w_qkv[l] = base;          w_out[l] = base + 12288;
        w_ff1[l] = base + 16384;  w_ff2[l] = base + 24576;
        cvt((const float*)d_in[LBASE[l] + 0], w_qkv[l], 192 * 64);
        cvt((const float*)d_in[LBASE[l] + 2], w_out[l], 64 * 64);
        cvt((const float*)d_in[LBASE[l] + 8], w_ff1[l], 128 * 64);
        cvt((const float*)d_in[LBASE[l] + 10], w_ff2[l], 64 * 128);
    }
    _Float16* w_pkv = w16 + 2 * 32768;             // pool qkv rows 64..191 (K,V)
    cvt(p_qkvw + 64 * 64, w_pkv, 128 * 64);

    // --- embedding ---
    drugseq_kernel<<<(BATCH * EMB + 255) / 256, 256, 0, stream>>>(drug_ids, drug_emb, dsp_w, dseq);
    embed_kernel<<<(BS * EMB + 255) / 256, 256, 0, stream>>>(tokens, positions, nt_embed, pos_embed, dseq, e0);

    // --- conv stack ---
    conv1_kernel<<<BS, 64, 0, stream>>>(e0, conv1_w, ln1_g, ln1_b, c1);
    conv2_kernel<<<BS, 64, 0, stream>>>(c1, conv2_w, ln2_g, ln2_b, x);

    // --- encoder layers ---
    const dim3 gLN(BS / 8), bLN(256);
    const dim3 bW(32);
    for (int l = 0; l < 2; l++) {
        const float* qkv_b = (const float*)d_in[LBASE[l] + 1];
        const float* out_b = (const float*)d_in[LBASE[l] + 3];
        const float* n1g   = (const float*)d_in[LBASE[l] + 4];
        const float* n1b   = (const float*)d_in[LBASE[l] + 5];
        const float* n2g   = (const float*)d_in[LBASE[l] + 6];
        const float* n2b   = (const float*)d_in[LBASE[l] + 7];
        const float* ff1_b = (const float*)d_in[LBASE[l] + 9];
        const float* ff2_b = (const float*)d_in[LBASE[l] + 11];

        // h = LN1(x)  -> f16
        ln_f16_kernel<<<gLN, bLN, 0, stream>>>(x, n1g, n1b, h16, BS);
        // qkv = h @ Wqkv^T + b   [BS,192] f16
        gemm_wmma_kernel<<<dim3(BS / 16, 192 / 16), bW, 0, stream>>>(
            h16, w_qkv[l], qkv_b, nullptr, nullptr, b16, 192, 64, GF_F16OUT);
        // windowed attention -> a16 [BS,64]
        win_attn_kernel<<<(BS * NHEAD + 255) / 256, 256, 0, stream>>>(b16, a16);
        // x = x + attn @ Wo^T + bo
        gemm_wmma_kernel<<<dim3(BS / 16, 64 / 16), bW, 0, stream>>>(
            a16, w_out[l], out_b, x, x, nullptr, 64, 64, GF_RESID);
        // h = LN2(x) -> f16
        ln_f16_kernel<<<gLN, bLN, 0, stream>>>(x, n2g, n2b, h16, BS);
        // g = gelu(h @ Wff1^T + b) -> f16 [BS,128]
        gemm_wmma_kernel<<<dim3(BS / 16, 128 / 16), bW, 0, stream>>>(
            h16, w_ff1[l], ff1_b, nullptr, nullptr, b16, 128, 64, GF_F16OUT | GF_GELU);
        // x = x + g @ Wff2^T + b
        gemm_wmma_kernel<<<dim3(BS / 16, 64 / 16), bW, 0, stream>>>(
            b16, w_ff2[l], ff2_b, x, x, nullptr, 64, 128, GF_RESID);
    }

    // --- pooling cross-attention ---
    cvt_f16_kernel<<<(BS * 64 + 255) / 256, 256, 0, stream>>>(x, x16, BS * 64);
    // kv = x @ Wkv^T + bkv   [BS,128] f16
    gemm_wmma_kernel<<<dim3(BS / 16, 128 / 16), bW, 0, stream>>>(
        x16, w_pkv, p_qkvb + 64, nullptr, nullptr, b16, 128, 64, GF_F16OUT);
    // q projection (tiny, f32)
    poolq_kernel<<<(BATCH * 64 + 255) / 256, 256, 0, stream>>>(
        drug_ids, drug_emb, p_baseq, p_dpw, p_qkvw, p_qkvb, qp);
    // softmax over S per (b,h)
    pool_attn_kernel<<<BATCH * NHEAD, 256, 0, stream>>>(qp, b16, pool);
    // out-proj + LN + MLP head
    head_kernel<<<BATCH, 64, 0, stream>>>(pool, p_outw, p_outb, p_lng, p_lnb,
                                          h_h1w, h_h1b, h_h2w, h_h2b, y_out);
}
